// Finder_21792664060770
// MI455X (gfx1250) — compile-verified, matrix operand order checked by reading, not run on hardware
//
#include <hip/hip_runtime.h>
#include <hip/hip_bf16.h>
#include <cfloat>

#define N_NODES 20000
#define N_EDGES 320000
#define FDIM    200
#define CHN     200
#define FP      224   // padded K dim (7 chunks of 32)
#define CP      208   // padded output dim (13 tiles of 16)
#define KC      7     // K chunks for F-dim GEMMs
#define INV_SQRT_DK 0.07071067811865475f
#define INV_BN      1.4128011f        // 1/sqrt(0.5 + 1e-3)

typedef __attribute__((ext_vector_type(16))) __bf16 v16bf;
typedef __attribute__((ext_vector_type(8)))  __bf16 v8bf;
typedef __attribute__((ext_vector_type(8)))  float  v8f;

// ---------------- WMMA helpers (CDNA5 16x16x32 bf16) ----------------
// A/B fragment: lane holds row (lane&15); k-chunks of 8 elems at
// kb=(lane>>4)*8 and kb+16 within the 32-wide K chunk (per ISA 7.12.2).
__device__ __forceinline__ v16bf frag_from(const __bf16* rowBase, int lane) {
  int kb = (lane >> 4) << 3;
  v8bf lo = *(const v8bf*)(rowBase + kb);
  v8bf hi = *(const v8bf*)(rowBase + kb + 16);
  v16bf f;
#pragma unroll
  for (int t = 0; t < 8; ++t) { f[t] = lo[t]; f[t + 8] = hi[t]; }
  return f;
}

// A fragment = elementwise average of two rows (nm = (x_i + x_j)/2),
// computed on the fly from L2-resident bf16 x.
__device__ __forceinline__ v16bf frag_avg(const __bf16* ra, const __bf16* rb, int lane) {
  int kb = (lane >> 4) << 3;
  v8bf a0 = *(const v8bf*)(ra + kb);
  v8bf a1 = *(const v8bf*)(ra + kb + 16);
  v8bf b0 = *(const v8bf*)(rb + kb);
  v8bf b1 = *(const v8bf*)(rb + kb + 16);
  v16bf f;
#pragma unroll
  for (int t = 0; t < 8; ++t) {
    f[t]     = (__bf16)(0.5f * ((float)a0[t] + (float)b0[t]));
    f[t + 8] = (__bf16)(0.5f * ((float)a1[t] + (float)b1[t]));
  }
  return f;
}

__device__ __forceinline__ v8f wmma_bf16(v16bf a, v16bf b, v8f c) {
  return __builtin_amdgcn_wmma_f32_16x16x32_bf16(false, a, false, b, (short)0, c,
                                                 false, false);
}

__device__ __forceinline__ void atomicMaxFloat(float* addr, float val) {
  if (val >= 0.0f) atomicMax((int*)addr, __float_as_int(val));
  else             atomicMin((unsigned int*)addr, __float_as_uint(val));
}

// ---------------- utility kernels ----------------
__global__ void k_fill(float* p, long long n, float v) {
  long long i = (long long)blockIdx.x * blockDim.x + threadIdx.x;
  if (i < n) p[i] = v;
}

__global__ void k_xconv(const float* __restrict__ x, __bf16* __restrict__ xb) {
  int n = blockIdx.x, c = threadIdx.x;
  if (c < FP) {
    float v = (c < FDIM) ? x[(size_t)n * FDIM + c] : 0.0f;
    xb[(size_t)n * FP + c] = (__bf16)v;
  }
}

// Wt[n*KP + k] = bf16(W[(k+rowOff)*ld + n]) with zero pad. grid = NP blocks.
__global__ void k_wt(const float* __restrict__ W, __bf16* __restrict__ Wt,
                     int K, int N, int KP, int rowOff, int ld) {
  int n = blockIdx.x, k = threadIdx.x;
  if (k < KP) {
    float v = (k < K && n < N) ? W[(size_t)(k + rowOff) * ld + n] : 0.0f;
    Wt[(size_t)n * KP + k] = (__bf16)v;
  }
}

// ---------------- fused node projections: Q,K,V,I share the A operand ----
// One wave per (row-tile, col-tile); A fragments preloaded once in registers
// and reused across the 4 weight streams (4x FLOP per byte of A traffic).
__global__ __launch_bounds__(128)
void k_node_qkvi(const __bf16* __restrict__ A,
                 const __bf16* __restrict__ Wqt, const __bf16* __restrict__ Wkt,
                 const __bf16* __restrict__ Wvt, const __bf16* __restrict__ Wit,
                 const float* __restrict__ bq, const float* __restrict__ bk,
                 const float* __restrict__ bv, const float* __restrict__ bi,
                 float* __restrict__ Qn, float* __restrict__ Kn,
                 float* __restrict__ Vn, float* __restrict__ In) {
  const int Nt = CP / 16;
  int wave = blockIdx.x * (blockDim.x >> 5) + (threadIdx.x >> 5);
  int lane = threadIdx.x & 31;
  if (wave >= (N_NODES / 16) * Nt) return;   // whole-wave uniform exit
  int rt = wave / Nt, ct = wave % Nt;
  const __bf16* arow = A + (size_t)(rt * 16 + (lane & 15)) * FP;
  v16bf af[KC];
#pragma unroll
  for (int kc = 0; kc < KC; ++kc) af[kc] = frag_from(arow + kc * 32, lane);

  int cl = lane & 15, mb = (lane >> 4) * 8;
  int c = ct * 16 + cl;
  const __bf16* wrow[4] = {
      Wqt + (size_t)c * FP, Wkt + (size_t)c * FP,
      Wvt + (size_t)c * FP, Wit + (size_t)c * FP };
  const float* bias[4] = { bq, bk, bv, bi };
  float* out[4] = { Qn, Kn, Vn, In };
#pragma unroll
  for (int t = 0; t < 4; ++t) {
    v8f acc = {};
#pragma unroll
    for (int kc = 0; kc < KC; ++kc)
      acc = wmma_bf16(af[kc], frag_from(wrow[t] + kc * 32, lane), acc);
    float bvv = (c < CHN) ? bias[t][c] : 0.0f;
#pragma unroll
    for (int v = 0; v < 8; ++v) {
      float val = acc[v] + bvv;
      if (t == 3) val = fmaxf(val, 0.0f);   // internal projection has relu
      out[t][(size_t)(rt * 16 + mb + v) * CP + c] = val;
    }
  }
}

// ---------------- gattr scatter (scatter_mean numerator) ----------------
__global__ void k_gattr(const float* __restrict__ x, const int* __restrict__ ii,
                        const int* __restrict__ jj, float* __restrict__ gattr,
                        float* __restrict__ cnt) {
  int e = blockIdx.x;
  int i = ii[e], j = jj[e];
  int c = threadIdx.x;
  if (c < FDIM) {
    float nm = 0.5f * (x[(size_t)i * FDIM + c] + x[(size_t)j * FDIM + c]);
    atomicAdd(&gattr[(size_t)i * CP + c], nm);
  }
  if (c == 0) atomicAdd(&cnt[i], 1.0f);
}

__global__ void k_gnorm(const float* __restrict__ gattr, const float* __restrict__ cnt,
                        __bf16* __restrict__ gbf) {
  int n = blockIdx.x, c = threadIdx.x;
  if (c < FP) {
    float v = (c < FDIM) ? gattr[(size_t)n * CP + c] / fmaxf(cnt[n], 1.0f) : 0.0f;
    gbf[(size_t)n * FP + c] = (__bf16)v;
  }
}

// ---------------- fused EdgeNet: (nm|g_at) -> 128 -> 64 -> 1 ----------------
// One wave per 16-edge tile; hidden activations live in LDS; A fragments
// (nm average + gathered gattr) preloaded once and reused over column tiles.
__global__ __launch_bounds__(128)
void k_edgenet(const int* __restrict__ idx_i, const int* __restrict__ idx_j,
               const __bf16* __restrict__ xb, const __bf16* __restrict__ gbf,
               const __bf16* __restrict__ We1at, const __bf16* __restrict__ We1bt,
               const float* __restrict__ be1,
               const __bf16* __restrict__ We2t, const float* __restrict__ be2,
               const float* __restrict__ We3, const float* __restrict__ be3,
               float* __restrict__ edge_out) {
  __shared__ __bf16 sh1[4][16 * 128];
  __shared__ __bf16 sh2[4][16 * 64];
  __shared__ int    sii[4][16], sjj[4][16];
  int w = threadIdx.x >> 5, lane = threadIdx.x & 31;
  int e0 = (blockIdx.x * 4 + w) * 16;
  if (lane < 16) { sii[w][lane] = idx_i[e0 + lane]; sjj[w][lane] = idx_j[e0 + lane]; }
  __syncthreads();
  int r = lane & 15, cl = lane & 15, mb = (lane >> 4) * 8;
  const __bf16* rowI = xb  + (size_t)sii[w][r] * FP;
  const __bf16* rowJ = xb  + (size_t)sjj[w][r] * FP;
  const __bf16* rowG = gbf + (size_t)sii[w][r] * FP;
  v16bf anm[KC], ag[KC];
#pragma unroll
  for (int kc = 0; kc < KC; ++kc) {
    anm[kc] = frag_avg(rowI + kc * 32, rowJ + kc * 32, lane);
    ag[kc]  = frag_from(rowG + kc * 32, lane);
  }
  // layer1: 2*F -> 128 as two accumulated K=224 GEMMs
  for (int ct = 0; ct < 8; ++ct) {
    v8f acc = {};
    const __bf16* b1 = We1at + (size_t)(ct * 16 + cl) * FP;
    const __bf16* b2 = We1bt + (size_t)(ct * 16 + cl) * FP;
#pragma unroll
    for (int kc = 0; kc < KC; ++kc)
      acc = wmma_bf16(anm[kc], frag_from(b1 + kc * 32, lane), acc);
#pragma unroll
    for (int kc = 0; kc < KC; ++kc)
      acc = wmma_bf16(ag[kc], frag_from(b2 + kc * 32, lane), acc);
    float bv = be1[ct * 16 + cl];
#pragma unroll
    for (int v = 0; v < 8; ++v)
      sh1[w][(mb + v) * 128 + ct * 16 + cl] = (__bf16)fmaxf(acc[v] + bv, 0.0f);
  }
  __syncthreads();
  // layer2: 128 -> 64 (A fragments from LDS, preloaded)
  {
    const __bf16* a2 = &sh1[w][r * 128];
    v16bf a2f[4];
#pragma unroll
    for (int kc = 0; kc < 4; ++kc) a2f[kc] = frag_from(a2 + kc * 32, lane);
    for (int ct = 0; ct < 4; ++ct) {
      v8f acc = {};
      const __bf16* b = We2t + (size_t)(ct * 16 + cl) * 128;
#pragma unroll
      for (int kc = 0; kc < 4; ++kc)
        acc = wmma_bf16(a2f[kc], frag_from(b + kc * 32, lane), acc);
      float bv = be2[ct * 16 + cl];
#pragma unroll
      for (int v = 0; v < 8; ++v)
        sh2[w][(mb + v) * 64 + ct * 16 + cl] = (__bf16)fmaxf(acc[v] + bv, 0.0f);
    }
  }
  __syncthreads();
  // layer3: 64 -> 1 dot product per edge
  if (lane < 16) {
    float acc = be3[0];
    const __bf16* h = &sh2[w][lane * 64];
#pragma unroll
    for (int k = 0; k < 64; ++k) acc += (float)h[k] * We3[k];
    edge_out[e0 + lane] = acc;
  }
}

// ---------------- segment softmax statistics ----------------
__global__ void k_smax(const int* __restrict__ ii, const int* __restrict__ jj,
                       const float* __restrict__ Qn, const float* __restrict__ Kn,
                       float* __restrict__ smax) {
  int e = blockIdx.x, c = threadIdx.x;
  if (c < CHN) {
    int i = ii[e], j = jj[e];
    float s = Qn[(size_t)i * CP + c] * Kn[(size_t)j * CP + c] * INV_SQRT_DK;
    atomicMaxFloat(&smax[(size_t)i * CP + c], s);
  }
}

__global__ void k_denom(const int* __restrict__ ii, const int* __restrict__ jj,
                        const float* __restrict__ Qn, const float* __restrict__ Kn,
                        const float* __restrict__ smax, float* __restrict__ den) {
  int e = blockIdx.x, c = threadIdx.x;
  if (c < CHN) {
    int i = ii[e], j = jj[e];
    float s = Qn[(size_t)i * CP + c] * Kn[(size_t)j * CP + c] * INV_SQRT_DK;
    atomicAdd(&den[(size_t)i * CP + c], __expf(s - smax[(size_t)i * CP + c]));
  }
}

// ---------------- fused SNet + attention + scatter ----------------
__global__ __launch_bounds__(128)
void k_snet(const int* __restrict__ idx_i, const int* __restrict__ idx_j,
            const __bf16* __restrict__ xb,
            const __bf16* __restrict__ Ws1at, const float* __restrict__ ws1_last,
            const float* __restrict__ bs1,
            const __bf16* __restrict__ Ws2t, const float* __restrict__ bs2,
            const __bf16* __restrict__ Ws3t, const float* __restrict__ bs3,
            const float* __restrict__ edge_g,
            const float* __restrict__ Qn, const float* __restrict__ Kn,
            const float* __restrict__ Vn,
            const float* __restrict__ smax, const float* __restrict__ den,
            float* __restrict__ emb) {
  __shared__ __bf16 sh1[4][16 * 128];
  __shared__ __bf16 sh2[4][16 * 64];
  __shared__ int    sii[4][16], sjj[4][16];
  __shared__ float  sedge[4][16];
  int w = threadIdx.x >> 5, lane = threadIdx.x & 31;
  int e0 = (blockIdx.x * 4 + w) * 16;
  if (lane < 16) {
    sii[w][lane] = idx_i[e0 + lane];
    sjj[w][lane] = idx_j[e0 + lane];
    sedge[w][lane] = edge_g[e0 + lane];
  }
  __syncthreads();
  int r = lane & 15, cl = lane & 15, mb = (lane >> 4) * 8;
  const __bf16* rowI = xb + (size_t)sii[w][r] * FP;
  const __bf16* rowJ = xb + (size_t)sjj[w][r] * FP;
  v16bf anm[KC];
#pragma unroll
  for (int kc = 0; kc < KC; ++kc)
    anm[kc] = frag_avg(rowI + kc * 32, rowJ + kc * 32, lane);
  // layer1: (F+1) -> 128 as K=224 GEMM + rank-1 edge term
  for (int ct = 0; ct < 8; ++ct) {
    v8f acc = {};
    const __bf16* b = Ws1at + (size_t)(ct * 16 + cl) * FP;
#pragma unroll
    for (int kc = 0; kc < KC; ++kc)
      acc = wmma_bf16(anm[kc], frag_from(b + kc * 32, lane), acc);
    int c = ct * 16 + cl;
    float wl = ws1_last[c], bv = bs1[c];
#pragma unroll
    for (int v = 0; v < 8; ++v) {
      float val = acc[v] + sedge[w][mb + v] * wl + bv;
      sh1[w][(mb + v) * 128 + c] = (__bf16)fmaxf(val, 0.0f);
    }
  }
  __syncthreads();
  // layer2: 128 -> 64
  {
    const __bf16* a2 = &sh1[w][r * 128];
    v16bf a2f[4];
#pragma unroll
    for (int kc = 0; kc < 4; ++kc) a2f[kc] = frag_from(a2 + kc * 32, lane);
    for (int ct = 0; ct < 4; ++ct) {
      v8f acc = {};
      const __bf16* b = Ws2t + (size_t)(ct * 16 + cl) * 128;
#pragma unroll
      for (int kc = 0; kc < 4; ++kc)
        acc = wmma_bf16(a2f[kc], frag_from(b + kc * 32, lane), acc);
      float bv = bs2[ct * 16 + cl];
#pragma unroll
      for (int v = 0; v < 8; ++v)
        sh2[w][(mb + v) * 64 + ct * 16 + cl] = (__bf16)fmaxf(acc[v] + bv, 0.0f);
    }
  }
  __syncthreads();
  // layer3: 64 -> 200, fused attention weight + scatter into emb
  {
    const __bf16* a3 = &sh2[w][r * 64];
    v16bf a3f[2];
#pragma unroll
    for (int kc = 0; kc < 2; ++kc) a3f[kc] = frag_from(a3 + kc * 32, lane);
    for (int ct = 0; ct < 13; ++ct) {
      v8f acc = {};
      const __bf16* b = Ws3t + (size_t)(ct * 16 + cl) * 64;
#pragma unroll
      for (int kc = 0; kc < 2; ++kc)
        acc = wmma_bf16(a3f[kc], frag_from(b + kc * 32, lane), acc);
      int c = ct * 16 + cl;
      if (c < CHN) {
        float bv = bs3[c];
#pragma unroll
        for (int v = 0; v < 8; ++v) {
          int m = mb + v;
          int i = sii[w][m], j = sjj[w][m];
          size_t oi = (size_t)i * CP + c, oj = (size_t)j * CP + c;
          float s   = Qn[oi] * Kn[oj] * INV_SQRT_DK;
          float ex  = __expf(s - smax[oi]);
          float aij = ex / (den[oi] + 1e-9f) * Vn[oj];
          float msg = aij * (acc[v] + bv) * INV_BN;
          atomicAdd(&emb[oi], msg);
        }
      }
    }
  }
}

// ---------------- final node update ----------------
__global__ void k_out(const float* __restrict__ In, const float* __restrict__ emb,
                      const float* __restrict__ cnt, float* __restrict__ out) {
  int n = blockIdx.x, c = threadIdx.x;
  if (c < CHN) {
    float v = In[(size_t)n * CP + c] + emb[(size_t)n * CP + c] / fmaxf(cnt[n], 1.0f);
    out[(size_t)n * CHN + c] = fmaxf(v, 0.0f);
  }
}

extern "C" void kernel_launch(void* const* d_in, const int* in_sizes, int n_in,
                              void* d_out, int out_size, void* d_ws, size_t ws_size,
                              hipStream_t stream) {
  (void)in_sizes; (void)n_in; (void)out_size; (void)ws_size;
  const float* x     = (const float*)d_in[0];
  const int*   idx_i = (const int*)d_in[1];
  const int*   idx_j = (const int*)d_in[2];
  const float* Wq = (const float*)d_in[3];  const float* bq = (const float*)d_in[4];
  const float* Wk = (const float*)d_in[5];  const float* bk = (const float*)d_in[6];
  const float* Wv = (const float*)d_in[7];  const float* bv = (const float*)d_in[8];
  const float* We1 = (const float*)d_in[9];  const float* be1 = (const float*)d_in[10];
  const float* We2 = (const float*)d_in[11]; const float* be2 = (const float*)d_in[12];
  const float* We3 = (const float*)d_in[13]; const float* be3 = (const float*)d_in[14];
  const float* Ws1 = (const float*)d_in[15]; const float* bs1 = (const float*)d_in[16];
  const float* Ws2 = (const float*)d_in[17]; const float* bs2 = (const float*)d_in[18];
  const float* Ws3 = (const float*)d_in[19]; const float* bs3 = (const float*)d_in[20];
  const float* Wi  = (const float*)d_in[21]; const float* bi  = (const float*)d_in[22];

  char* ws = (char*)d_ws;
  size_t off = 0;
  auto alloc = [&](size_t bytes) -> void* {
    void* p = ws + off;
    off = (off + bytes + 255) & ~(size_t)255;
    return p;
  };
  __bf16* xb    = (__bf16*)alloc((size_t)N_NODES * FP * 2);
  __bf16* gbf   = (__bf16*)alloc((size_t)N_NODES * FP * 2);
  __bf16* Wqt   = (__bf16*)alloc((size_t)CP * FP * 2);
  __bf16* Wkt   = (__bf16*)alloc((size_t)CP * FP * 2);
  __bf16* Wvt   = (__bf16*)alloc((size_t)CP * FP * 2);
  __bf16* Wit   = (__bf16*)alloc((size_t)CP * FP * 2);
  __bf16* We1at = (__bf16*)alloc((size_t)128 * FP * 2);
  __bf16* We1bt = (__bf16*)alloc((size_t)128 * FP * 2);
  __bf16* We2t  = (__bf16*)alloc((size_t)64 * 128 * 2);
  __bf16* Ws1at = (__bf16*)alloc((size_t)128 * FP * 2);
  __bf16* Ws2t  = (__bf16*)alloc((size_t)64 * 128 * 2);
  __bf16* Ws3t  = (__bf16*)alloc((size_t)CP * 64 * 2);
  float* Qn    = (float*)alloc((size_t)N_NODES * CP * 4);
  float* Kn    = (float*)alloc((size_t)N_NODES * CP * 4);
  float* Vn    = (float*)alloc((size_t)N_NODES * CP * 4);
  float* In    = (float*)alloc((size_t)N_NODES * CP * 4);
  float* gattr = (float*)alloc((size_t)N_NODES * CP * 4);
  float* smaxb = (float*)alloc((size_t)N_NODES * CP * 4);
  float* denb  = (float*)alloc((size_t)N_NODES * CP * 4);
  float* embb  = (float*)alloc((size_t)N_NODES * CP * 4);
  float* cnt   = (float*)alloc((size_t)N_NODES * 4);
  float* edge_g= (float*)alloc((size_t)N_EDGES * 4);

  long long ncp = (long long)N_NODES * CP;
  int fb = (int)((ncp + 255) / 256);
  k_fill<<<fb, 256, 0, stream>>>(gattr, ncp, 0.0f);
  k_fill<<<fb, 256, 0, stream>>>(smaxb, ncp, -FLT_MAX);
  k_fill<<<fb, 256, 0, stream>>>(denb,  ncp, 0.0f);
  k_fill<<<fb, 256, 0, stream>>>(embb,  ncp, 0.0f);
  k_fill<<<(N_NODES + 255) / 256, 256, 0, stream>>>(cnt, N_NODES, 0.0f);

  k_xconv<<<N_NODES, 256, 0, stream>>>(x, xb);
  k_wt<<<CP, 256, 0, stream>>>(Wq, Wqt, FDIM, CHN, FP, 0, CHN);
  k_wt<<<CP, 256, 0, stream>>>(Wk, Wkt, FDIM, CHN, FP, 0, CHN);
  k_wt<<<CP, 256, 0, stream>>>(Wv, Wvt, FDIM, CHN, FP, 0, CHN);
  k_wt<<<CP, 256, 0, stream>>>(Wi, Wit, FDIM, CHN, FP, 0, CHN);
  k_wt<<<128, 256, 0, stream>>>(We1, We1at, FDIM, 128, FP, 0, 128);
  k_wt<<<128, 256, 0, stream>>>(We1, We1bt, FDIM, 128, FP, FDIM, 128);
  k_wt<<<64, 256, 0, stream>>>(We2, We2t, 128, 64, 128, 0, 64);
  k_wt<<<128, 256, 0, stream>>>(Ws1, Ws1at, FDIM, 128, FP, 0, 128);
  k_wt<<<64, 256, 0, stream>>>(Ws2, Ws2t, 128, 64, 128, 0, 64);
  k_wt<<<CP, 256, 0, stream>>>(Ws3, Ws3t, 64, CHN, 64, 0, CHN);

  // fused node projections: 1250 row-tiles x 13 col-tiles, 4 waves/block
  int tiles = (N_NODES / 16) * (CP / 16);
  int ngb = (tiles + 3) / 4;
  k_node_qkvi<<<ngb, 128, 0, stream>>>(xb, Wqt, Wkt, Wvt, Wit,
                                       bq, bk, bv, bi, Qn, Kn, Vn, In);

  k_gattr<<<N_EDGES, 256, 0, stream>>>(x, idx_i, idx_j, gattr, cnt);
  k_gnorm<<<N_NODES, 256, 0, stream>>>(gattr, cnt, gbf);

  int egb = N_EDGES / 16 / 4;   // 5000 blocks, 4 waves each, 16 edges/wave
  k_edgenet<<<egb, 128, 0, stream>>>(idx_i, idx_j, xb, gbf, We1at, We1bt, be1,
                                     We2t, be2, We3, be3, edge_g);

  k_smax<<<N_EDGES, 256, 0, stream>>>(idx_i, idx_j, Qn, Kn, smaxb);
  k_denom<<<N_EDGES, 256, 0, stream>>>(idx_i, idx_j, Qn, Kn, smaxb, denb);

  const float* ws1_last = Ws1 + (size_t)FDIM * 128;   // row 200 of [201,128]
  k_snet<<<egb, 128, 0, stream>>>(idx_i, idx_j, xb, Ws1at, ws1_last, bs1,
                                  Ws2t, bs2, Ws3t, bs3, edge_g,
                                  Qn, Kn, Vn, smaxb, denb, embb);

  k_out<<<N_NODES, 256, 0, stream>>>(In, embb, cnt, (float*)d_out);
}
